// OctreeAttention_84078279787152
// MI455X (gfx1250) — matrix operand order repaired and checked by reading, not compiled, not run on hardware
//
#include <hip/hip_runtime.h>
#include <hip/hip_bf16.h>

// ---------------------------------------------------------------------------
// OctreeT windowed attention for MI455X (gfx1250, wave32, WMMA).
//   pass 0: one-shot fp32 -> bf16 conversion of activations + weights
//   pass 1: QKV GEMM   (bf16 x bf16 -> f32) via v_wmma_f32_16x16x32_bf16
//   pass 2: windowed attention (fp32 VALU, LDS-staged), writes bf16
//   pass 3: proj GEMM  (bf16 x bf16 -> f32) via v_wmma_f32_16x16x32_bf16
// GEMM: A panel register-resident (read from HBM once); B double-buffered in
// registers so next tile's loads overlap current tile's WMMA chain.
// ---------------------------------------------------------------------------

typedef __attribute__((ext_vector_type(16))) __bf16 v16bf;
typedef __attribute__((ext_vector_type(8)))  __bf16 v8bf;
typedef __attribute__((ext_vector_type(4)))  __bf16 v4bf;
typedef __attribute__((ext_vector_type(8)))  float  v8f;

#define NTOK   192000
#define CDIM   256
#define HH     8
#define KK     24
#define DD     4
#define WW     8000      // NTOK / KK
#define CH     32        // CDIM / HH
#define POSB   38
#define RPEN   77
#define NEGV   (-1000.0f)

// dilation permutation: window w, slot j -> original row
__device__ __forceinline__ int dil_row(int w, int j) {
  return (w >> 2) * (KK * DD) + j * DD + (w & 3);
}

// ---------------------------------------------------------------------------
// One-shot fp32 -> bf16 (RNE via hardware cvt). n must be a multiple of 4.
// ---------------------------------------------------------------------------
__global__ void __launch_bounds__(256)
cvt_f32_bf16(const float* __restrict__ in, __bf16* __restrict__ out, int n) {
  const int i = ((int)blockIdx.x * 256 + (int)threadIdx.x) * 4;
  if (i < n) {
    const float4 v = *(const float4*)(in + i);
    v4bf o;
    o[0] = (__bf16)v.x; o[1] = (__bf16)v.y;
    o[2] = (__bf16)v.z; o[3] = (__bf16)v.w;
    *(v4bf*)(out + i) = o;
  }
}

// ---------------------------------------------------------------------------
// B-fragment tile loader: 8 x v16bf (one 16x16x32 B fragment per K-step).
//   B layout (ISA 7.12.2): lane L: N = L&15; elems 0..7 -> K = 8*(L>>4)+0..7 ; 8..15 -> +16
// ---------------------------------------------------------------------------
__device__ __forceinline__ void load_btile(v16bf (&b)[8], const __bf16* bp) {
#pragma unroll
  for (int ks = 0; ks < 8; ++ks) {
    const v8bf blo = *(const v8bf*)(bp + ks * 32);
    const v8bf bhi = *(const v8bf*)(bp + ks * 32 + 16);
    b[ks] = __builtin_shufflevector(blo, bhi, 0, 1, 2, 3, 4, 5, 6, 7,
                                    8, 9, 10, 11, 12, 13, 14, 15);
  }
}

// ---------------------------------------------------------------------------
// Y[M x NOUT] = X[M x 256](bf16) @ Wm[NOUT x 256](bf16)^T + bias, f32 out.
// grid = (M/128), block = 256 (8 waves). Wave w owns rows [bx*128+w*16, +16).
// A fragments (full K=256) live in 64 VGPRs for the whole kernel; B tiles are
// double-buffered in registers (b0/b1) so the 16 b128 loads of tile t+1 are
// in flight while the 8-WMMA chain of tile t executes.
// Fragment layouts per CDNA5 ISA 7.12.2 (16x16x32 bf16):
//   A: lane L: M = L&15; elems 0..7  -> K = 8*(L>>4)+0..7 ; 8..15 -> +16
//   C/D: lane L: N = L&15; vgpr r -> M = r + 8*(L>>4)
// ---------------------------------------------------------------------------
template <int NOUT>
__global__ void __launch_bounds__(256)
gemm_bf16_wmma(const __bf16* __restrict__ X, const __bf16* __restrict__ Wm,
               const float* __restrict__ bias, float* __restrict__ Y) {
  const int wave  = threadIdx.x >> 5;
  const int lane  = threadIdx.x & 31;
  const int half  = lane >> 4;
  const int l16   = lane & 15;
  const int mBase = (int)blockIdx.x * 128 + wave * 16;
  const int row   = mBase + l16;

  // Preload the wave's entire A panel-row into registers (64 VGPRs).
  const __bf16* __restrict__ ap = X + (size_t)row * CDIM;
  v16bf afr[8];
#pragma unroll
  for (int ks = 0; ks < 8; ++ks) {
    const v8bf alo = *(const v8bf*)(ap + ks * 32 + half * 8);
    const v8bf ahi = *(const v8bf*)(ap + ks * 32 + 16 + half * 8);
    afr[ks] = __builtin_shufflevector(alo, ahi, 0, 1, 2, 3, 4, 5, 6, 7,
                                      8, 9, 10, 11, 12, 13, 14, 15);
  }

  constexpr int nTiles = NOUT >> 4;                 // even (48 or 16)
  const __bf16* __restrict__ wbase = Wm + (size_t)l16 * CDIM + half * 8;
  const size_t tileStride = (size_t)16 * CDIM;      // 16 cols of W

  v16bf b0[8], b1[8];
  load_btile(b0, wbase);

  for (int ot = 0; ot < nTiles; ot += 2) {
    // kick off loads for tile ot+1 while computing tile ot
    load_btile(b1, wbase + (size_t)(ot + 1) * tileStride);
    if (ot + 2 < nTiles)
      __builtin_prefetch(wbase + (size_t)(ot + 2) * tileStride, 0, 1);

    v8f acc = {};
#pragma unroll
    for (int ks = 0; ks < 8; ++ks)
      acc = __builtin_amdgcn_wmma_f32_16x16x32_bf16(
          false, afr[ks], false, b0[ks], (short)0, acc, false, false);
    {
      const int col = (ot << 4) + l16;
      const float bv = bias[col];
#pragma unroll
      for (int r = 0; r < 8; ++r)
        Y[(size_t)(mBase + half * 8 + r) * NOUT + col] = acc[r] + bv;
    }

    // kick off loads for tile ot+2 while computing tile ot+1
    if (ot + 2 < nTiles)
      load_btile(b0, wbase + (size_t)(ot + 2) * tileStride);

    acc = (v8f){};
#pragma unroll
    for (int ks = 0; ks < 8; ++ks)
      acc = __builtin_amdgcn_wmma_f32_16x16x32_bf16(
          false, afr[ks], false, b1[ks], (short)0, acc, false, false);
    {
      const int col = ((ot + 1) << 4) + l16;
      const float bv = bias[col];
#pragma unroll
      for (int r = 0; r < 8; ++r)
        Y[(size_t)(mBase + half * 8 + r) * NOUT + col] = acc[r] + bv;
    }
  }
}

// ---------------------------------------------------------------------------
// Attention: one block per window (grid = 8000), one wave per head.
// Stages the window's 24x768 qkv slice in LDS, then lane j (< 24) owns query
// j: scores (QK^T * scale) + RPE gather + dilate/causal mask -> softmax ->
// P@V. Output stored as bf16 in un-dilated row order for the proj GEMM.
// ---------------------------------------------------------------------------
__global__ void __launch_bounds__(256)
attn_kernel(const float* __restrict__ qkv, const float* __restrict__ rpe,
            const int* __restrict__ dpos, const float* __restrict__ dmask,
            __bf16* __restrict__ attout) {
  __shared__ float smem[KK * 3 * CDIM];   // 24 x 768 fp32 = 73728 B
  const int w   = blockIdx.x;
  const int tid = threadIdx.x;

  for (int idx = tid; idx < KK * 3 * CDIM; idx += 256) {
    const int j = idx / (3 * CDIM);
    const int c = idx - j * (3 * CDIM);
    smem[idx] = qkv[(size_t)dil_row(w, j) * (3 * CDIM) + c];
  }
  __syncthreads();

  const int h    = tid >> 5;   // head = wave index
  const int lane = tid & 31;   // query slot
  if (lane < KK) {
    const int j = lane;
    const float scale = 0.17677669529663689f;  // 1/sqrt(32)

    float qreg[CH];
#pragma unroll
    for (int c = 0; c < CH; ++c)
      qreg[c] = smem[j * (3 * CDIM) + h * CH + c];

    float s[KK];
#pragma unroll
    for (int i = 0; i < KK; ++i) {
      const float* kp = &smem[i * (3 * CDIM) + CDIM + h * CH];
      float acc = 0.f;
#pragma unroll
      for (int c = 0; c < CH; ++c) acc += qreg[c] * kp[c];
      acc *= scale;

      // RPE bias: sum over 3 axes of rpe[(clip(p)+38 + 77*a) * H + h]
      const int* pp = dpos + (((size_t)w * KK + j) * KK + i) * 3;
      float bias = 0.f;
#pragma unroll
      for (int a = 0; a < 3; ++a) {
        int p = pp[a];
        p = (p < -POSB) ? -POSB : ((p > POSB) ? POSB : p);
        bias += rpe[(p + POSB + a * RPEN) * HH + h];
      }
      const float m = (i <= j) ? dmask[((size_t)w * KK + j) * KK + i] : NEGV;
      s[i] = acc + bias + m;
    }

    // softmax over 24 keys, in registers
    float mx = s[0];
#pragma unroll
    for (int i = 1; i < KK; ++i) mx = fmaxf(mx, s[i]);
    float sum = 0.f;
#pragma unroll
    for (int i = 0; i < KK; ++i) { s[i] = __expf(s[i] - mx); sum += s[i]; }
    const float inv = __frcp_rn(sum);

    float o[CH];
#pragma unroll
    for (int c = 0; c < CH; ++c) o[c] = 0.f;
#pragma unroll
    for (int i = 0; i < KK; ++i) {
      const float p = s[i] * inv;
      const float* vp = &smem[i * (3 * CDIM) + 2 * CDIM + h * CH];
#pragma unroll
      for (int c = 0; c < CH; ++c) o[c] += p * vp[c];
    }

    __bf16* op = attout + (size_t)dil_row(w, j) * CDIM + h * CH;
#pragma unroll
    for (int c = 0; c < CH; c += 4) {
      v4bf ob;
      ob[0] = (__bf16)o[c];     ob[1] = (__bf16)o[c + 1];
      ob[2] = (__bf16)o[c + 2]; ob[3] = (__bf16)o[c + 3];
      *(v4bf*)(op + c) = ob;
    }
  }
}

// ---------------------------------------------------------------------------
extern "C" void kernel_launch(void* const* d_in, const int* in_sizes, int n_in,
                              void* d_out, int out_size, void* d_ws, size_t ws_size,
                              hipStream_t stream) {
  const float* data   = (const float*)d_in[0];   // [N, 256]
  const float* qkv_w  = (const float*)d_in[1];   // [768, 256]
  const float* qkv_b  = (const float*)d_in[2];   // [768]
  const float* proj_w = (const float*)d_in[3];   // [256, 256]
  const float* proj_b = (const float*)d_in[4];   // [256]
  const float* rpe    = (const float*)d_in[5];   // [231, 8]
  const int*   dpos   = (const int*)d_in[6];     // [W, 24, 24, 3]
  const float* dmask  = (const float*)d_in[7];   // [W, 24, 24]
  float* out = (float*)d_out;                    // [N, 256] fp32

  // workspace layout (4-byte types first)
  char* ws = (char*)d_ws;
  float* qkv = (float*)ws;            ws += (size_t)NTOK * (3 * CDIM) * 4;
  __bf16* xb     = (__bf16*)ws;       ws += (size_t)NTOK * CDIM * 2;
  __bf16* attout = (__bf16*)ws;       ws += (size_t)NTOK * CDIM * 2;
  __bf16* wqb    = (__bf16*)ws;       ws += (size_t)(3 * CDIM) * CDIM * 2;
  __bf16* wpb    = (__bf16*)ws;       ws += (size_t)CDIM * CDIM * 2;

  dim3 blk(256);
  // pass 0: convert activations + weights to bf16 (one touch each)
  cvt_f32_bf16<<<dim3((NTOK * CDIM) / 1024), blk, 0, stream>>>(
      data, xb, NTOK * CDIM);
  cvt_f32_bf16<<<dim3((3 * CDIM * CDIM) / 1024), blk, 0, stream>>>(
      qkv_w, wqb, 3 * CDIM * CDIM);
  cvt_f32_bf16<<<dim3((CDIM * CDIM) / 1024), blk, 0, stream>>>(
      proj_w, wpb, CDIM * CDIM);

  // pass 1: QKV projection  [N,256] x [256,768] -> f32
  gemm_bf16_wmma<3 * CDIM><<<dim3(NTOK / 128), blk, 0, stream>>>(
      xb, wqb, qkv_b, qkv);
  // pass 2: windowed attention (8000 windows x 8 heads) -> bf16
  attn_kernel<<<dim3(WW), blk, 0, stream>>>(qkv, rpe, dpos, dmask, attout);
  // pass 3: output projection  [N,256] x [256,256] -> f32
  gemm_bf16_wmma<CDIM><<<dim3(NTOK / 128), blk, 0, stream>>>(
      attout, wpb, proj_b, out);
}
